// HardAttentionHead_1743756722644
// MI455X (gfx1250) — matrix operand order repaired and checked by reading, not compiled, not run on hardware
//
#include <hip/hip_runtime.h>

// ---------------------------------------------------------------------------
// HardAttentionHead for MI455X (gfx1250, wave32, WMMA 16x16x32 bf16)
//   x [4,4096,1024] f32 ; Wq/Wk/Wv [1024,128] f32
//   out = (att @ v) [4,4096,128] f32 ; att [4,4096,4096] f32 (softmax probs)
//   d_out = out ++ att (flat, return order)
//   d_ws  = q,k,v in bf16 (3 * 16384*128*2 = 12.6 MB; assumed available)
// ---------------------------------------------------------------------------

#define NB   4
#define TT   4096
#define DIN  1024
#define DD   128
#define ROWS (NB * TT)          // 16384
#define ATT_STRIDE 4100         // 4096 + pad (avoid 64-bank conflicts)
#define Q_STRIDE   130          // 128 + pad (even -> dword-pair aligned)

typedef __attribute__((ext_vector_type(16))) __bf16 v16bf;
typedef __attribute__((ext_vector_type(8)))  float  v8f;

union AFrag { v16bf bf; unsigned int u[8]; };

// A-matrix (16-bit, 16x32 MxK) lane/VGPR K-base per ISA 7.12.2:
// lanes 0-15: V0..3 -> K=0,2,4,6 ; V4..7 -> K=16,18,20,22 ; lanes 16-31: +8
__device__ __forceinline__ int kbaseA(int lane, int v) {
    return (v < 4 ? 2 * v : 2 * v + 8) + ((lane & 16) ? 8 : 0);
}
// B-matrix (16-bit, 32x16 KxN): same K order as A (V0..3 -> K=0,2,4,6 ; V4..7 -> K=16..22 ; lanes 16-31: +8)
__device__ __forceinline__ int kbaseB(int lane, int v) {
    return (v < 4 ? 2 * v : 2 * v + 8) + ((lane & 16) ? 8 : 0);
}

__device__ __forceinline__ unsigned int pack_bf2(float lo, float hi) {
    union { __bf16 b[2]; unsigned int u; } t;
    t.b[0] = (__bf16)lo; t.b[1] = (__bf16)hi;
    return t.u;
}

// ---------------------------------------------------------------------------
// Kernel 1: q/k/v = x @ W   (bf16 WMMA, f32 accumulate, bf16 store to ws)
// grid: (ROWS/16, 3)  block: 256 (8 waves, each owns a 16-col stripe)
// ---------------------------------------------------------------------------
__global__ __launch_bounds__(256) void qkv_proj_kernel(
    const float* __restrict__ x,
    const float* __restrict__ Wq, const float* __restrict__ Wk,
    const float* __restrict__ Wv,
    __bf16* __restrict__ qkv)
{
    const int lane = threadIdx.x & 31;
    const int wave = threadIdx.x >> 5;
    const int row0 = blockIdx.x * 16;
    const float* W = (blockIdx.y == 0) ? Wq : (blockIdx.y == 1 ? Wk : Wv);
    __bf16* out = qkv + (size_t)blockIdx.y * ((size_t)ROWS * DD);

    const int d0 = wave * 16;
    const int mA = lane & 15;   // A row
    const int nn = lane & 15;   // B/C col
    v8f c = {};

    for (int c0 = 0; c0 < DIN; c0 += 32) {
        AFrag a, b;
#pragma unroll
        for (int v = 0; v < 8; ++v) {
            const int ka = c0 + kbaseA(lane, v);
            const float* xp = x + (size_t)(row0 + mA) * DIN + ka;
            a.u[v] = pack_bf2(xp[0], xp[1]);
            const int kb = c0 + kbaseB(lane, v);
            const float* wp = W + (size_t)kb * DD + d0 + nn;
            b.u[v] = pack_bf2(wp[0], wp[DD]);
        }
        c = __builtin_amdgcn_wmma_f32_16x16x32_bf16(
                false, a.bf, false, b.bf, (short)0, c, false, false);
    }

    const int mC = (lane & 16) ? 8 : 0;
#pragma unroll
    for (int v = 0; v < 8; ++v)
        out[(size_t)(row0 + mC + v) * DD + d0 + nn] = (__bf16)c[v];
}

// ---------------------------------------------------------------------------
// Kernel 2: fused causal attention for one 16-row strip.
// grid: (TT/16, NB)  block: 256 (8 waves)
// LDS: att strip [16][ATT_STRIDE] f32 (262 KB) + q tile [16][Q_STRIDE] bf16
// ---------------------------------------------------------------------------
__global__ __launch_bounds__(256) void attn_kernel(
    const __bf16* __restrict__ qkv,
    float* __restrict__ outO,   // [NB,TT,DD]
    float* __restrict__ attO)   // [NB,TT,TT]
{
    extern __shared__ char smem[];
    float*  attS = (float*)smem;                                  // [16][ATT_STRIDE]
    __bf16* qS   = (__bf16*)(smem + 16 * ATT_STRIDE * sizeof(float)); // [16][Q_STRIDE]

    const int tid  = threadIdx.x;
    const int lane = tid & 31;
    const int wave = tid >> 5;
    const int n    = blockIdx.y;
    const int row0 = blockIdx.x * 16;
    const int ncols   = row0 + 16;                 // causal: cols [0, ncols)
    const int ncols32 = (ncols + 31) & ~31;        // round to WMMA K multiple

    const __bf16* qB = qkv;
    const __bf16* kB = qkv + (size_t)ROWS * DD;
    const __bf16* vB = qkv + 2ull * ROWS * DD;
    const size_t rowbase = (size_t)n * TT + row0;

    // stage q tile (bf16) into LDS
    for (int i = tid; i < 16 * DD; i += 256) {
        const int m = i >> 7, d = i & (DD - 1);
        qS[m * Q_STRIDE + d] = qB[(rowbase + m) * DD + d];
    }
    // zero the K-padding column tile so att@v K-loop reads zeros there
    if (ncols32 > ncols) {
        const int m = tid >> 4, s = ncols + (tid & 15);
        attS[m * ATT_STRIDE + s] = 0.0f;
    }
    __syncthreads();

    const int mA = lane & 15;
    const int nn = lane & 15;
    const int mC = (lane & 16) ? 8 : 0;
    const float NEG_INF = -__builtin_inff();

    // ---- scores: S = q @ k^T, masked, into LDS ----
    for (int j = wave; j <= (int)blockIdx.x; j += 8) {
        const int s0 = j * 16;
        v8f c = {};
#pragma unroll
        for (int kc = 0; kc < 4; ++kc) {
            AFrag a, b;
            const unsigned int* q32 = (const unsigned int*)qS;
#pragma unroll
            for (int v = 0; v < 8; ++v) {
                const int ka = kc * 32 + kbaseA(lane, v);
                a.u[v] = q32[(mA * Q_STRIDE + ka) >> 1];
                const int kb = kc * 32 + kbaseB(lane, v);
                b.u[v] = *(const unsigned int*)
                    (kB + ((size_t)n * TT + s0 + nn) * DD + kb);
            }
            c = __builtin_amdgcn_wmma_f32_16x16x32_bf16(
                    false, a.bf, false, b.bf, (short)0, c, false, false);
        }
#pragma unroll
        for (int v = 0; v < 8; ++v) {
            const int m = mC + v;
            const int t = row0 + m;
            const int s = s0 + nn;
            float val = c[v];
            // faithful: tril zeroes upper triangle, then (att==0) -> -inf
            val = (s <= t && val != 0.0f) ? val : NEG_INF;
            attS[m * ATT_STRIDE + s] = val;
        }
    }
    __syncthreads();

    // ---- exact softmax per row (16 threads/row), write att to global ----
    {
        const int m  = tid >> 4;
        const int lp = tid & 15;
        const int t  = row0 + m;
        float* rowp = attS + m * ATT_STRIDE;

        float mx = NEG_INF;
        for (int s = lp; s < ncols; s += 16) mx = fmaxf(mx, rowp[s]);
#pragma unroll
        for (int o = 8; o >= 1; o >>= 1) mx = fmaxf(mx, __shfl_xor(mx, o, 16));

        float sum = 0.0f;
        for (int s = lp; s < ncols; s += 16) {
            const float e = __expf(rowp[s] - mx);
            rowp[s] = e;
            sum += e;
        }
#pragma unroll
        for (int o = 8; o >= 1; o >>= 1) sum += __shfl_xor(sum, o, 16);

        const float inv = 1.0f / sum;
        float* aOut = attO + ((size_t)n * TT + t) * TT;
        for (int s = lp; s < ncols; s += 16) {
            const float p = rowp[s] * inv;
            rowp[s] = p;
            aOut[s] = p;
        }
        for (int s = ncols + lp; s < TT; s += 16) aOut[s] = 0.0f; // upper triangle
    }
    __syncthreads();

    // ---- out = att @ v  (A = probs from LDS f32->bf16, B = v bf16) ----
    const int d0 = wave * 16;
    v8f c2 = {};
    const int nk = ncols32 >> 5;
    for (int kc = 0; kc < nk; ++kc) {
        AFrag a, b;
#pragma unroll
        for (int v = 0; v < 8; ++v) {
            const int ka = kc * 32 + kbaseA(lane, v);
            a.u[v] = pack_bf2(attS[mA * ATT_STRIDE + ka],
                              attS[mA * ATT_STRIDE + ka + 1]);
            const int kb = kc * 32 + kbaseB(lane, v);
            const __bf16* vp = vB + ((size_t)n * TT + kb) * DD + d0 + nn;
            union { __bf16 b2[2]; unsigned int u; } t2;
            t2.b2[0] = vp[0]; t2.b2[1] = vp[DD];
            b.u[v] = t2.u;
        }
        c2 = __builtin_amdgcn_wmma_f32_16x16x32_bf16(
                false, a.bf, false, b.bf, (short)0, c2, false, false);
    }
#pragma unroll
    for (int v = 0; v < 8; ++v)
        outO[(rowbase + mC + v) * DD + d0 + nn] = c2[v];
}

// ---------------------------------------------------------------------------
extern "C" void kernel_launch(void* const* d_in, const int* in_sizes, int n_in,
                              void* d_out, int out_size, void* d_ws, size_t ws_size,
                              hipStream_t stream) {
    const float* x  = (const float*)d_in[0];
    const float* Wq = (const float*)d_in[1];
    const float* Wk = (const float*)d_in[2];
    const float* Wv = (const float*)d_in[3];

    float* out  = (float*)d_out;                       // [4,4096,128]
    float* attO = out + (size_t)NB * TT * DD;          // [4,4096,4096]
    __bf16* qkv = (__bf16*)d_ws;                       // 12.6 MB bf16 q,k,v

    // 1) QKV projections (bf16 WMMA)
    qkv_proj_kernel<<<dim3(ROWS / 16, 3), 256, 0, stream>>>(x, Wq, Wk, Wv, qkv);

    // 2) fused causal attention, one workgroup per 16-row strip
    const size_t shmem = (size_t)16 * ATT_STRIDE * sizeof(float)
                       + (size_t)16 * Q_STRIDE * sizeof(__bf16);   // ~266 KB < 320 KB
    attn_kernel<<<dim3(TT / 16, NB), 256, shmem, stream>>>(qkv, out, attO);
}
